// SelfAttention_65146063946246
// MI455X (gfx1250) — matrix operand order repaired and checked by reading, not compiled, not run on hardware
//
#include <hip/hip_runtime.h>
#include <cstdint>

// ---------------------------------------------------------------------------
// CDNA5 (gfx1250) fused causal multi-head self-attention
//   pass 0a: transpose + convert Wq/Wk/Wv/Wo  f32[D,D] -> f16 Wt[N,K]
//   pass 0b: convert x -> f16
//   pass 1 : QKV = x@W+b  (async-LDS staged A, 128-wide K chunks)
//   pass 2 : flash attention per (b,h,16-query tile), online softmax
//   pass 3 : out = att@Wo + bo  (async-LDS staged A; f32 output)
// Matrix math: v_wmma_f32_16x16x32_f16 (wave32). A tiles: double-buffered LDS
// via global_load_async_to_lds_b128 (ASYNCcnt), one sync per 4 K-steps.
// ---------------------------------------------------------------------------

typedef __attribute__((ext_vector_type(16))) _Float16 v16h;
typedef __attribute__((ext_vector_type(4)))  _Float16 v4h;
typedef __attribute__((ext_vector_type(8)))  float    v8f;
typedef __attribute__((ext_vector_type(4)))  float    v4f;

#define WMMA16(acc, a, b) \
  __builtin_amdgcn_wmma_f32_16x16x32_f16(false, (a), false, (b), (short)0, (acc), false, false)

namespace {

constexpr int   kBatch  = 2;
constexpr int   kS      = 2048;
constexpr int   kD      = 1024;
constexpr int   kH      = 16;
constexpr int   kDh     = 64;
constexpr int   kNT     = kBatch * kS;    // 4096 tokens
constexpr int   kKB     = kD / 128;       // 8 staged K-chunks of 128
constexpr float kScale  = 0.03125f;       // 1/sqrt(1024) (ref divides by sqrt(embed_dim))
constexpr float kNegBig = -1.0e30f;

union FragU { v16h h; v4f f[2]; };

// A fragment (16x32 f16) from a row-major f16 matrix: two b128 loads per lane.
// Lane layout (CDNA5 ISA): m = lane%16, half = lane/16,
//   elem i(0..7): K = half*8+i ; elem 8+i: K = 16+half*8+i
__device__ __forceinline__ v16h load_a_f16(const _Float16* __restrict__ A, int ld,
                                           int row0, int k0) {
  const int lane = threadIdx.x & 31;
  const int half = lane >> 4;
  const int m    = lane & 15;
  const _Float16* base = A + (size_t)(row0 + m) * ld + k0 + half * 8;
  FragU u;
  u.f[0] = *reinterpret_cast<const v4f*>(base);
  u.f[1] = *reinterpret_cast<const v4f*>(base + 16);
  return u.h;
}

// A fragment from an LDS-resident 32x128 tile (row stride 128 halfs).
__device__ __forceinline__ v16h load_a_lds(const _Float16* buf, int row0loc, int k0loc) {
  const int lane = threadIdx.x & 31;
  const int half = lane >> 4;
  const int m    = lane & 15;
  const _Float16* base = buf + (row0loc + m) * 128 + k0loc + half * 8;
  FragU u;
  u.f[0] = *reinterpret_cast<const v4f*>(base);
  u.f[1] = *reinterpret_cast<const v4f*>(base + 16);
  return u.h;
}

// B fragment (32x16 f16) where the contract dim runs along the f16 row:
//   B[kc,n] = M[(row0+n)*ld + c0 + kc]
// (transposed weights Wt[N,K], K tiles of K[S,Dh], Vt[Dh,S]).
// Lane layout: n = lane%16, khalf = lane/16, elem i(0..7): Kc = khalf*8+i ; elem 8+i: Kc = 16+khalf*8+i.
__device__ __forceinline__ v16h load_bT_f16(const _Float16* __restrict__ M, int ld,
                                            int row0, int c0) {
  const int lane  = threadIdx.x & 31;
  const int n     = lane & 15;
  const int khalf = lane >> 4;
  const _Float16* base = M + (size_t)(row0 + n) * ld + c0 + khalf * 8;
  FragU u;
  u.f[0] = *reinterpret_cast<const v4f*>(base);
  u.f[1] = *reinterpret_cast<const v4f*>(base + 16);
  return u.h;
}

// One 16-byte async global->LDS transfer (per lane).
// LDS aperture: flat addr bits [31:0] are the DS address (ISA 10.2).
__device__ __forceinline__ void async_b128(const _Float16* gsrc, _Float16* lds_dst) {
  const uint32_t lds = (uint32_t)(uintptr_t)lds_dst;
  asm volatile("global_load_async_to_lds_b128 %0, %1, off"
               :: "v"(lds), "v"(gsrc) : "memory");
}

// Stage one 32x128 f16 A tile into LDS: 8KB, 256 threads x 2 x 16B.
__device__ __forceinline__ void stage_a_tile(const _Float16* __restrict__ src, int ld,
                                             int row0, int k0, _Float16* buf) {
  const int tid = threadIdx.x;
  const int row = tid >> 3;         // 0..31
  const int c8  = (tid & 7) * 8;    // halfs, 0..56
  const _Float16* g0 = src + (size_t)(row0 + row) * ld + k0 + c8;
  _Float16* l0 = buf + row * 128 + c8;
  async_b128(g0, l0);
  async_b128(g0 + 64, l0 + 64);
}

__device__ __forceinline__ void wait_async(int n) {
  if (n == 0) asm volatile("s_wait_asynccnt 0x0" ::: "memory");
  else        asm volatile("s_wait_asynccnt 0x2" ::: "memory");
}

} // namespace

// ---------------------------------------------------------------------------
// Pass 0a: LDS-tiled transpose + f32->f16 convert of the four weight matrices.
//   Wt[n*D + k] = (f16) W[k*D + n]
// ---------------------------------------------------------------------------
__global__ __launch_bounds__(256) void wconv_kernel(
    const float* __restrict__ Wq, const float* __restrict__ Wk,
    const float* __restrict__ Wv, const float* __restrict__ Wo,
    _Float16* __restrict__ wqt, _Float16* __restrict__ wkt,
    _Float16* __restrict__ wvt, _Float16* __restrict__ wot) {
  __shared__ _Float16 t[32][33];          // +1 pad: no LDS bank conflicts
  const int g   = blockIdx.x;             // 4 mats x 1024 tiles
  const int mat = g >> 10;
  const int ti  = g & 1023;
  const int tr  = ti >> 5, tc = ti & 31;
  const float*  W  = (mat == 0) ? Wq : (mat == 1) ? Wk : (mat == 2) ? Wv : Wo;
  _Float16*     Wt = (mat == 0) ? wqt : (mat == 1) ? wkt : (mat == 2) ? wvt : wot;
  const int lane = threadIdx.x & 31;
  const int w    = threadIdx.x >> 5;
#pragma unroll
  for (int i = 0; i < 4; ++i) {
    const int r = w + i * 8;
    t[r][lane] = (_Float16)W[(size_t)(tr * 32 + r) * kD + tc * 32 + lane];
  }
  __syncthreads();
#pragma unroll
  for (int i = 0; i < 4; ++i) {
    const int r = w + i * 8;
    Wt[(size_t)(tc * 32 + r) * kD + tr * 32 + lane] = t[lane][r];
  }
}

// ---------------------------------------------------------------------------
// Pass 0b: elementwise x f32 -> f16 (4 elems / thread, vectorized).
// ---------------------------------------------------------------------------
__global__ __launch_bounds__(256) void xconv_kernel(const float* __restrict__ x,
                                                    _Float16* __restrict__ x16) {
  const size_t i = ((size_t)blockIdx.x * 256 + threadIdx.x) * 4;
  const v4f v = *reinterpret_cast<const v4f*>(x + i);
  v4h h;
#pragma unroll
  for (int j = 0; j < 4; ++j) h[j] = (_Float16)v[j];
  *reinterpret_cast<v4h*>(x16 + i) = h;
}

// ---------------------------------------------------------------------------
// Pass 1: fused QKV projection. 8 waves/block share one 32-token row tile;
// a 32x128 A chunk is double-buffered in LDS via async-to-LDS copies, and each
// staged chunk feeds a barrier-free unrolled region of 4 K-steps (48 WMMAs).
// V is written transposed per head: Vt[B,H,Dh,S].
// ---------------------------------------------------------------------------
__global__ __launch_bounds__(256) void qkv_proj_kernel(
    const _Float16* __restrict__ x16,
    const _Float16* __restrict__ wqt, const float* __restrict__ bq,
    const _Float16* __restrict__ wkt, const float* __restrict__ bk,
    const _Float16* __restrict__ wvt, const float* __restrict__ bv,
    _Float16* __restrict__ qws, _Float16* __restrict__ kws,
    _Float16* __restrict__ vtws) {
  __shared__ alignas(16) _Float16 abuf[2][32 * 128];  // 2 x 8KB A-chunk buffers

  const int tid  = threadIdx.x;
  const int wave = tid >> 5;
  const int lane = tid & 31;
  const int g    = blockIdx.x * 8 + wave;  // 4096 wave-tiles
  const int rowt = g >> 5;                 // block-uniform (8 | 32 col tiles)
  const int colt = g & 31;
  const int row0 = rowt * 32, col0 = colt * 32;

  const _Float16* const Wts[3] = {wqt, wkt, wvt};
  const float* const    Bs[3]  = {bq, bk, bv};

  stage_a_tile(x16, kD, row0, 0, abuf[0]);

  v8f acc[3][2][2] = {};
  for (int kb = 0; kb < kKB; ++kb) {
    const int cur = kb & 1;
    if (kb + 1 < kKB) {
      stage_a_tile(x16, kD, row0, (kb + 1) * 128, abuf[cur ^ 1]);
      wait_async(2);  // chunk kb (our 2 asyncs) complete
    } else {
      wait_async(0);
    }
    __syncthreads();  // chunk kb complete for all waves

#pragma unroll
    for (int mat = 0; mat < 3; ++mat)
      __builtin_prefetch(Wts[mat] + (size_t)(col0 + (lane & 15)) * kD + (kb + 1) * 128, 0, 1);

#pragma unroll
    for (int s = 0; s < 4; ++s) {
      const int k0 = kb * 128 + s * 32;
      const v16h a0 = load_a_lds(abuf[cur], 0, s * 32);
      const v16h a1 = load_a_lds(abuf[cur], 16, s * 32);
#pragma unroll
      for (int mat = 0; mat < 3; ++mat) {
#pragma unroll
        for (int c = 0; c < 2; ++c) {
          const v16h bf = load_bT_f16(Wts[mat], kD, col0 + c * 16, k0);
          acc[mat][0][c] = WMMA16(acc[mat][0][c], a0, bf);
          acc[mat][1][c] = WMMA16(acc[mat][1][c], a1, bf);
        }
      }
    }
    asm volatile("s_wait_dscnt 0x0" ::: "memory");  // our LDS reads done
    __syncthreads();  // all waves done with chunk kb before it is overwritten
  }

  // C layout: row = (lane/16)*8 + r, col = lane%16
  const int half = lane >> 4, n = lane & 15;
#pragma unroll
  for (int mat = 0; mat < 3; ++mat) {
#pragma unroll
    for (int c = 0; c < 2; ++c) {
      const int col = col0 + c * 16 + n;
      const int h = col >> 6, dh = col & (kDh - 1);
      const float vb = Bs[mat][col];
#pragma unroll
      for (int rt = 0; rt < 2; ++rt) {
#pragma unroll
        for (int r = 0; r < 8; ++r) {
          const int row = row0 + rt * 16 + half * 8 + r;
          const int b = row >> 11, s = row & (kS - 1);
          const _Float16 val = (_Float16)(acc[mat][rt][c][r] + vb);
          if (mat == 0) {
            qws[(((size_t)(b * kH + h)) * kS + s) * kDh + dh] = val;
          } else if (mat == 1) {
            kws[(((size_t)(b * kH + h)) * kS + s) * kDh + dh] = val;
          } else {  // V stored transposed: [B,H,Dh,S]
            vtws[(((size_t)(b * kH + h)) * kDh + dh) * kS + s] = val;
          }
        }
      }
    }
  }
}

// ---------------------------------------------------------------------------
// Pass 2: flash attention. One wave owns a 16-query tile of one (b,h); loops
// over 32-key blocks (skipping fully-masked blocks via the causal bound).
// ---------------------------------------------------------------------------
__global__ __launch_bounds__(256) void flash_attn_kernel(
    const _Float16* __restrict__ qws, const _Float16* __restrict__ kws,
    const _Float16* __restrict__ vtws, _Float16* __restrict__ aws) {
  __shared__ alignas(16) _Float16 psm[8][16 * 32];  // per-wave P staging (8 KB)

  const int wave = threadIdx.x >> 5;
  const int lane = threadIdx.x & 31;
  const int half = lane >> 4, n = lane & 15;

  const int g  = blockIdx.x * 8 + wave;  // 4096 wave-tiles
  const int bh = g >> 7;
  const int qt = g & 127;
  const int b  = bh >> 4, h = bh & 15;

  const size_t head = (size_t)(b * kH + h) * kS * kDh;
  const _Float16* Qh = qws + head;
  const _Float16* Kh = kws + head;
  const _Float16* Vt = vtws + head;      // [Dh, S]

  const int q0 = qt * 16;

  // Q tile (16x64) kept in registers as two A fragments
  const v16h qa0 = load_a_f16(Qh, kDh, q0, 0);
  const v16h qa1 = load_a_f16(Qh, kDh, q0, 32);

  v8f o[4] = {{}, {}, {}, {}};   // 16x64 f32 output accumulator
  float mst[8], lst[8];
#pragma unroll
  for (int r = 0; r < 8; ++r) { mst[r] = kNegBig; lst[r] = 0.f; }

  const int nkb = (q0 + 16 + 31) >> 5;   // causal: only key blocks <= q tile
  for (int kb = 0; kb < nkb; ++kb) {
    const int key0 = kb * 32;

    // scores S = Q.K^T : 16x32 tile = two 16x16 C tiles, contract over Dh=64
    v8f s0 = {}, s1 = {};
    s0 = WMMA16(s0, qa0, load_bT_f16(Kh, kDh, key0, 0));
    s0 = WMMA16(s0, qa1, load_bT_f16(Kh, kDh, key0, 32));
    s1 = WMMA16(s1, qa0, load_bT_f16(Kh, kDh, key0 + 16, 0));
    s1 = WMMA16(s1, qa1, load_bT_f16(Kh, kDh, key0 + 16, 32));

    float alpha[8];
#pragma unroll
    for (int r = 0; r < 8; ++r) {
      const int row = q0 + half * 8 + r;  // global query index
      float v0 = s0[r] * kScale; if (key0 + n      > row) v0 = kNegBig;
      float v1 = s1[r] * kScale; if (key0 + 16 + n > row) v1 = kNegBig;
      // row max across the 16 lanes of this half-wave
      float mx = fmaxf(v0, v1);
      mx = fmaxf(mx, __shfl_xor(mx, 1));
      mx = fmaxf(mx, __shfl_xor(mx, 2));
      mx = fmaxf(mx, __shfl_xor(mx, 4));
      mx = fmaxf(mx, __shfl_xor(mx, 8));
      const float mnew = fmaxf(mst[r], mx);
      alpha[r] = __expf(mst[r] - mnew);
      const float p0 = __expf(v0 - mnew);
      const float p1 = __expf(v1 - mnew);
      float rs = p0 + p1;
      rs += __shfl_xor(rs, 1);
      rs += __shfl_xor(rs, 2);
      rs += __shfl_xor(rs, 4);
      rs += __shfl_xor(rs, 8);
      lst[r] = lst[r] * alpha[r] + rs;
      mst[r] = mnew;
      // stage probabilities through LDS (C-fragment -> A-fragment reshape)
      psm[wave][(half * 8 + r) * 32 + n]      = (_Float16)p0;
      psm[wave][(half * 8 + r) * 32 + 16 + n] = (_Float16)p1;
    }

    // rescale running output by alpha
#pragma unroll
    for (int c = 0; c < 4; ++c)
#pragma unroll
      for (int r = 0; r < 8; ++r) o[c][r] *= alpha[r];

    // wave-private LDS round trip (DS ops from one wave are in-order)
    asm volatile("s_wait_dscnt 0x0" ::: "memory");

    FragU u;
    const int m = lane & 15;
    u.f[0] = *reinterpret_cast<const v4f*>(&psm[wave][m * 32 + half * 8]);
    u.f[1] = *reinterpret_cast<const v4f*>(&psm[wave][m * 32 + 16 + half * 8]);
    const v16h pa = u.h;

    // O += P (16x32) . V (32x64); Vt rows are dh, cols are keys -> b128 loads
#pragma unroll
    for (int c = 0; c < 4; ++c)
      o[c] = WMMA16(o[c], pa, load_bT_f16(Vt, kS, c * 16, key0));
  }

  // normalize and store att output as f16 [B,S,D] (heads re-interleaved)
#pragma unroll
  for (int c = 0; c < 4; ++c) {
#pragma unroll
    for (int r = 0; r < 8; ++r) {
      const int row    = q0 + half * 8 + r;
      const size_t tok = (size_t)b * kS + row;
      const int col    = h * kDh + c * 16 + n;
      aws[tok * kD + col] = (_Float16)(o[c][r] / lst[r]);
    }
  }
}

// ---------------------------------------------------------------------------
// Pass 3: output projection (32x32 per wave), async-LDS staged A chunks,
// f32 epilogue into d_out.
// ---------------------------------------------------------------------------
__global__ __launch_bounds__(256) void out_proj_kernel(
    const _Float16* __restrict__ aws, const _Float16* __restrict__ wot,
    const float* __restrict__ bo, float* __restrict__ out) {
  __shared__ alignas(16) _Float16 abuf[2][32 * 128];

  const int tid  = threadIdx.x;
  const int wave = tid >> 5;
  const int lane = tid & 31;
  const int g    = blockIdx.x * 8 + wave;  // 4096 wave-tiles
  const int rowt = g >> 5, colt = g & 31;  // rowt block-uniform
  const int row0 = rowt * 32, col0 = colt * 32;

  stage_a_tile(aws, kD, row0, 0, abuf[0]);

  v8f acc[2][2] = {};
  for (int kb = 0; kb < kKB; ++kb) {
    const int cur = kb & 1;
    if (kb + 1 < kKB) {
      stage_a_tile(aws, kD, row0, (kb + 1) * 128, abuf[cur ^ 1]);
      wait_async(2);
    } else {
      wait_async(0);
    }
    __syncthreads();

    __builtin_prefetch(wot + (size_t)(col0 + (lane & 15)) * kD + (kb + 1) * 128, 0, 1);
#pragma unroll
    for (int s = 0; s < 4; ++s) {
      const int k0 = kb * 128 + s * 32;
      const v16h a0 = load_a_lds(abuf[cur], 0, s * 32);
      const v16h a1 = load_a_lds(abuf[cur], 16, s * 32);
#pragma unroll
      for (int c = 0; c < 2; ++c) {
        const v16h bf = load_bT_f16(wot, kD, col0 + c * 16, k0);
        acc[0][c] = WMMA16(acc[0][c], a0, bf);
        acc[1][c] = WMMA16(acc[1][c], a1, bf);
      }
    }
    asm volatile("s_wait_dscnt 0x0" ::: "memory");
    __syncthreads();
  }

  const int half = lane >> 4, n = lane & 15;
#pragma unroll
  for (int c = 0; c < 2; ++c) {
    const int col = col0 + c * 16 + n;
    const float vb = bo[col];
#pragma unroll
    for (int rt = 0; rt < 2; ++rt) {
#pragma unroll
      for (int r = 0; r < 8; ++r) {
        const int row = row0 + rt * 16 + half * 8 + r;
        out[(size_t)row * kD + col] = acc[rt][c][r] + vb;
      }
    }
  }
}

// ---------------------------------------------------------------------------
extern "C" void kernel_launch(void* const* d_in, const int* in_sizes, int n_in,
                              void* d_out, int out_size, void* d_ws, size_t ws_size,
                              hipStream_t stream) {
  (void)in_sizes; (void)n_in; (void)out_size; (void)ws_size;
  const float* x  = (const float*)d_in[0];
  const float* Wq = (const float*)d_in[1];
  const float* bq = (const float*)d_in[2];
  const float* Wk = (const float*)d_in[3];
  const float* bk = (const float*)d_in[4];
  const float* Wv = (const float*)d_in[5];
  const float* bv = (const float*)d_in[6];
  const float* Wo = (const float*)d_in[7];
  const float* bo = (const float*)d_in[8];

  // f16 workspace (48 MB total)
  _Float16* x16 = (_Float16*)d_ws;                       // [NT, D]
  _Float16* wqt = x16 + (size_t)kNT * kD;                // [D, D] transposed
  _Float16* wkt = wqt + (size_t)kD * kD;
  _Float16* wvt = wkt + (size_t)kD * kD;
  _Float16* wot = wvt + (size_t)kD * kD;
  _Float16* qws = wot + (size_t)kD * kD;                 // [B,H,S,Dh]
  _Float16* kws = qws + (size_t)kNT * kD;                // [B,H,S,Dh]
  _Float16* vtws = kws + (size_t)kNT * kD;               // [B,H,Dh,S]
  _Float16* aws = vtws + (size_t)kNT * kD;               // [B,S,D]

  wconv_kernel<<<4096, 256, 0, stream>>>(Wq, Wk, Wv, Wo, wqt, wkt, wvt, wot);
  xconv_kernel<<<4096, 256, 0, stream>>>(x, x16);
  qkv_proj_kernel<<<512, 256, 0, stream>>>(x16, wqt, bq, wkt, bk, wvt, bv,
                                           qws, kws, vtws);
  flash_attn_kernel<<<512, 256, 0, stream>>>(qws, kws, vtws, aws);
  out_proj_kernel<<<512, 256, 0, stream>>>(aws, wot, bo, (float*)d_out);
}
